// DipoleMomentDecoder_27625229648411
// MI455X (gfx1250) — matrix operand order repaired and checked by reading, not compiled, not run on hardware
//
#include <hip/hip_runtime.h>

// ---------------------------------------------------------------------------
// Types for CDNA5 WMMA (wave32): V_WMMA_F32_16X16X32_BF16
// ---------------------------------------------------------------------------
typedef __attribute__((ext_vector_type(16))) __bf16 v16bf;
typedef __attribute__((ext_vector_type(8)))  float  v8f;

union FragU {
    v16bf v;
    unsigned int u[8];
};

__device__ __forceinline__ unsigned short f2bf(float x) {
    union { float f; unsigned u; } v; v.f = x;
    unsigned r = v.u + 0x7FFFu + ((v.u >> 16) & 1u);   // round-to-nearest-even
    return (unsigned short)(r >> 16);
}
__device__ __forceinline__ float bf2f(unsigned short b) {
    union { unsigned u; float f; } v; v.u = ((unsigned)b) << 16;
    return v.f;
}
__device__ __forceinline__ float silu(float x) {
    return x / (1.0f + __expf(-x));
}

// A fragment (16x32 bf16, MxK) gathered from an LDS tile (row-major, stride in
// bf16 elements).  Contiguous k order (matches fragB_wt): lane m=l&15; pair p holds
// K = 16*(l>=16) + 2*p (+0/+1) -> contiguous pair = one b32.
__device__ __forceinline__ v16bf fragA_lds(const unsigned short* tile, int stride, int lane) {
    FragU f;
    const int m   = lane & 15;
    const int klo = (lane & 16) ? 16 : 0;
    const unsigned short* row = tile + m * stride;
#pragma unroll
    for (int p = 0; p < 8; ++p) {
        int k0 = klo + (p << 1);
        f.u[p] = *reinterpret_cast<const unsigned int*>(row + k0);
    }
    return f.v;
}

// B fragment (32x16 bf16, KxN) gathered from a transposed (OUT-major) weight
// matrix: element (k,n) lives at wT[n*ldk + k].  Contiguous k order (same as A): lane n=l&15,
// K-half = 16*(l>=16), pair p holds K = half + 2p (+0/+1) -> contiguous b32,
// so the 8 dwords form one contiguous 32B run -> two global_load_b128.
__device__ __forceinline__ v16bf fragB_wt(const unsigned short* __restrict__ wT,
                                          int ldk, int n0, int k0, int lane) {
    FragU f;
    const int n  = n0 + (lane & 15);
    const int kh = (lane & 16) ? 16 : 0;
    const unsigned short* row = wT + (long)n * ldk + k0 + kh;
#pragma unroll
    for (int p = 0; p < 8; ++p) {
        f.u[p] = *reinterpret_cast<const unsigned int*>(row + 2 * p);
    }
    return f.v;
}

#define WMMA_BF16(a, b, c) \
    __builtin_amdgcn_wmma_f32_16x16x32_bf16(false, (a), false, (b), (short)0, (c), false, false)

// ---------------------------------------------------------------------------
// Prologue: convert weights fp32 -> bf16, transposed to OUT-major
// ---------------------------------------------------------------------------
__global__ void conv_wT_kernel(const float* __restrict__ src, unsigned short* __restrict__ dstT,
                               int K, int Nc) {
    int i = blockIdx.x * blockDim.x + threadIdx.x;
    if (i < K * Nc) {
        int k = i / Nc, n = i % Nc;
        dstT[n * K + k] = f2bf(src[i]);
    }
}

// g1_w is 128x129: first 128 cols -> bf16 transposed, last col -> fp32 vector
__global__ void conv_g1w_kernel(const float* __restrict__ src, unsigned short* __restrict__ dstT,
                                float* __restrict__ wlast) {
    int i = blockIdx.x * blockDim.x + threadIdx.x;
    if (i < 128 * 129) {
        int k = i / 129, n = i % 129;
        if (n < 128) dstT[n * 128 + k] = f2bf(src[i]);
        else         wlast[k] = src[i];
    }
}

__global__ void zero_kernel(float* __restrict__ p, int n) {
    int i = blockIdx.x * blockDim.x + threadIdx.x;
    if (i < n) p[i] = 0.0f;
}

// ---------------------------------------------------------------------------
// Gate 0 (fused GEMM A + channel-norm + MLP GEMMs B,C + gating)
// One workgroup = 16 nodes, 8 waves (256 threads).
// ---------------------------------------------------------------------------
__global__ __launch_bounds__(256) void gate0_kernel(
    const float* __restrict__ vec,           // (N,3,256) fp32
    const float* __restrict__ scal,          // (N,256)   fp32
    const unsigned short* __restrict__ W0T,  // 384x256 bf16 (out-major)
    const unsigned short* __restrict__ M1T,  // 512x512 bf16
    const unsigned short* __restrict__ M2T,  // 256x512 bf16
    const float* __restrict__ b1,            // (512)
    const float* __restrict__ b2,            // (256)
    unsigned short* __restrict__ qws,        // (N,128) bf16 out
    unsigned short* __restrict__ muws)       // (N,3,128) bf16 out
{
    __shared__ union {
        unsigned short a[48 * 256];                               // GEMM A input tile (24KB)
        struct { unsigned short xb[16 * 512];                     // [scaler | vecnorm] (16KB)
                 unsigned short h[16 * 512]; } s;                 // silu(x@M1+b1)     (16KB)
    } U;
    __shared__ unsigned short sVw[48 * 128];                      // vec_w (bf16, 12KB)
    __shared__ float sNs[16 * 256];                               // norm^2 accum (16KB)

    const int tid  = threadIdx.x;
    const int lane = tid & 31;
    const int wv   = __builtin_amdgcn_readfirstlane((int)threadIdx.x) >> 5;  // scalar wave id
    const long node0 = (long)blockIdx.x * 16;

    // ---- stage vector tile (48 rows x 256) as bf16 (b128 reads, b64 writes)
    {
        const float4* src4 = reinterpret_cast<const float4*>(vec + node0 * 3 * 256);
        for (int i = tid; i < 48 * 256 / 4; i += 256) {
            float4 v = src4[i];
            unsigned lo = (unsigned)f2bf(v.x) | ((unsigned)f2bf(v.y) << 16);
            unsigned hi = (unsigned)f2bf(v.z) | ((unsigned)f2bf(v.w) << 16);
            *reinterpret_cast<uint2*>(U.a + 4 * i) = make_uint2(lo, hi);
        }
        for (int i = tid; i < 16 * 256; i += 256) sNs[i] = 0.0f;
    }
    __syncthreads();

    // ---- GEMM A: (48x256) @ (256x384); 24 col-tile jobs, B reused across the
    //      3 row-tiles (3 accumulators) -> 1 B-fragment per 3 WMMAs
    for (int nt = wv; nt < 24; nt += 8) {
        v8f acc0 = {}, acc1 = {}, acc2 = {};
#pragma unroll
        for (int kc = 0; kc < 8; ++kc) {
            v16bf b  = fragB_wt(W0T, 256, nt * 16, kc * 32, lane);
            v16bf a0 = fragA_lds(U.a            + kc * 32, 256, lane);
            v16bf a1 = fragA_lds(U.a + 16 * 256 + kc * 32, 256, lane);
            v16bf a2 = fragA_lds(U.a + 32 * 256 + kc * 32, 256, lane);
            acc0 = WMMA_BF16(a0, b, acc0);
            acc1 = WMMA_BF16(a1, b, acc1);
            acc2 = WMMA_BF16(a2, b, acc2);
        }
        const int n = lane & 15, hi = (lane & 16) ? 8 : 0;
        if (nt < 16) {                       // vec_v -> accumulate norm^2 per node
            const int col = nt * 16 + n;
#pragma unroll
            for (int r = 0; r < 8; ++r) {
                const int m = r + hi;        // row = node*3 + c
                atomicAdd(&sNs[((m)      / 3) * 256 + col], acc0[r] * acc0[r]);
                atomicAdd(&sNs[((16 + m) / 3) * 256 + col], acc1[r] * acc1[r]);
                atomicAdd(&sNs[((32 + m) / 3) * 256 + col], acc2[r] * acc2[r]);
            }
        } else {                             // vec_w -> keep in LDS for gating
            const int col = (nt - 16) * 16 + n;
#pragma unroll
            for (int r = 0; r < 8; ++r) {
                const int m = r + hi;
                sVw[(m)      * 128 + col] = f2bf(acc0[r]);
                sVw[(16 + m) * 128 + col] = f2bf(acc1[r]);
                sVw[(32 + m) * 128 + col] = f2bf(acc2[r]);
            }
        }
    }
    __syncthreads();

    // ---- build x = [scaler | sqrt(norm^2)] as bf16 (overwrites GEMM-A tile)
    {
        const float* sc = scal + node0 * 256;
        for (int i = tid; i < 16 * 512; i += 256) {
            const int nr = i >> 9, col = i & 511;
            float x = (col < 256) ? sc[nr * 256 + col]
                                  : sqrtf(sNs[nr * 256 + (col - 256)]);
            U.s.xb[i] = f2bf(x);
        }
    }
    __syncthreads();

    // ---- GEMM B: (16x512) @ (512x512), 4 col-tiles per wave sharing one
    //      A-fragment per k-step; bias + SiLU -> H (bf16 in LDS)
    {
        v8f acc[4] = {};
#pragma unroll
        for (int kc = 0; kc < 16; ++kc) {
            v16bf a = fragA_lds(U.s.xb + kc * 32, 512, lane);
#pragma unroll
            for (int t = 0; t < 4; ++t) {
                v16bf b = fragB_wt(M1T, 512, (wv + 8 * t) * 16, kc * 32, lane);
                acc[t] = WMMA_BF16(a, b, acc[t]);
            }
        }
        const int n = lane & 15, hi = (lane & 16) ? 8 : 0;
#pragma unroll
        for (int t = 0; t < 4; ++t) {
            const int col = (wv + 8 * t) * 16 + n;
            const float bb = b1[col];
#pragma unroll
            for (int r = 0; r < 8; ++r)
                U.s.h[(r + hi) * 512 + col] = f2bf(silu(acc[t][r] + bb));
        }
    }
    __syncthreads();

    // ---- GEMM C: (16x512) @ (512x256) + bias; wave wv owns q-tile wv and
    //      gate-tile wv (both accumulated off one shared A-fragment)
    {
        v8f accQ = {}, accG = {};
#pragma unroll
        for (int kc = 0; kc < 16; ++kc) {
            v16bf a  = fragA_lds(U.s.h + kc * 32, 512, lane);
            v16bf bq = fragB_wt(M2T, 512, wv * 16,       kc * 32, lane);
            v16bf bg = fragB_wt(M2T, 512, 128 + wv * 16, kc * 32, lane);
            accQ = WMMA_BF16(a, bq, accQ);
            accG = WMMA_BF16(a, bg, accG);
        }
        const int n = lane & 15, hi = (lane & 16) ? 8 : 0;
        const int colq = wv * 16 + n;                 // 0..127
        const float bbq = b2[colq];
        const float bbg = b2[128 + colq];
#pragma unroll
        for (int r = 0; r < 8; ++r) {
            const int  m    = r + hi;
            const long node = node0 + m;
            qws[node * 128 + colq] = f2bf(silu(accQ[r] + bbq));     // q = silu(s_out)
            const float g = accG[r] + bbg;                          // gate
#pragma unroll
            for (int c = 0; c < 3; ++c)
                muws[(node * 3 + c) * 128 + colq] =
                    f2bf(g * bf2f(sVw[(m * 3 + c) * 128 + colq]));
        }
    }
}

// ---------------------------------------------------------------------------
// Gate 1 (GEMM D + norm + MLP GEMMs E,F + node_mu + segment sum)
// ---------------------------------------------------------------------------
__global__ __launch_bounds__(256) void gate1_kernel(
    const unsigned short* __restrict__ muws,  // (N,3,128) bf16
    const unsigned short* __restrict__ qws,   // (N,128)   bf16
    const unsigned short* __restrict__ W1T,   // 128x128 bf16 (out-major)
    const float* __restrict__ w1last,         // (128) fp32, last col of g1_w
    const unsigned short* __restrict__ N1T,   // 256x256 bf16
    const float* __restrict__ b1,             // (256)
    const unsigned short* __restrict__ N2T,   // 2x256 bf16 (out-major)
    const float* __restrict__ b2,             // (2)
    const float* __restrict__ pos,            // (N,3)
    const float* __restrict__ mc,             // (B,3)
    const int* __restrict__ bidx,             // (N)
    float* __restrict__ gacc)                 // (B,3) accumulator
{
    __shared__ unsigned short sMu[48 * 128];  // 12KB
    __shared__ float sNs[16 * 128];           // 8KB
    __shared__ float sVw1[48];
    __shared__ unsigned short sXE[16 * 256];  // 8KB
    __shared__ unsigned short sH2[16 * 256];  // 8KB
    __shared__ float sX4[32];

    const int tid  = threadIdx.x;
    const int lane = tid & 31;
    const int wv   = __builtin_amdgcn_readfirstlane((int)threadIdx.x) >> 5;
    const long node0 = (long)blockIdx.x * 16;

    {
        const uint4* src4 = reinterpret_cast<const uint4*>(muws + node0 * 3 * 128);
        for (int i = tid; i < 48 * 128 / 8; i += 256)
            *reinterpret_cast<uint4*>(sMu + 8 * i) = src4[i];
        for (int i = tid; i < 16 * 128; i += 256) sNs[i] = 0.0f;
    }
    __syncthreads();

    // ---- GEMM D: (48x128) @ (128x128); wave wv owns col-tile wv, B reused
    //      across 3 row-tiles -> norm^2 accumulation
    {
        const int nt = wv;                    // 8 waves cover nt 0..7 exactly
        v8f acc0 = {}, acc1 = {}, acc2 = {};
#pragma unroll
        for (int kc = 0; kc < 4; ++kc) {
            v16bf b  = fragB_wt(W1T, 128, nt * 16, kc * 32, lane);
            v16bf a0 = fragA_lds(sMu            + kc * 32, 128, lane);
            v16bf a1 = fragA_lds(sMu + 16 * 128 + kc * 32, 128, lane);
            v16bf a2 = fragA_lds(sMu + 32 * 128 + kc * 32, 128, lane);
            acc0 = WMMA_BF16(a0, b, acc0);
            acc1 = WMMA_BF16(a1, b, acc1);
            acc2 = WMMA_BF16(a2, b, acc2);
        }
        const int n = lane & 15, hi = (lane & 16) ? 8 : 0;
        const int col = nt * 16 + n;
#pragma unroll
        for (int r = 0; r < 8; ++r) {
            const int m = r + hi;
            atomicAdd(&sNs[((m)      / 3) * 128 + col], acc0[r] * acc0[r]);
            atomicAdd(&sNs[((16 + m) / 3) * 128 + col], acc1[r] * acc1[r]);
            atomicAdd(&sNs[((32 + m) / 3) * 128 + col], acc2[r] * acc2[r]);
        }
    }
    __syncthreads();

    // ---- vec_w1 (extra 129th column of g1_w) + build x = [q | norm]
    if (tid < 48) {
        float d = 0.0f;
        for (int i = 0; i < 128; ++i) d += bf2f(sMu[tid * 128 + i]) * w1last[i];
        sVw1[tid] = d;
    }
    for (int i = tid; i < 16 * 256; i += 256) {
        const int nr = i >> 8, col = i & 255;
        sXE[i] = (col < 128) ? qws[(node0 + nr) * 128 + col]
                             : f2bf(sqrtf(sNs[nr * 128 + (col - 128)]));
    }
    __syncthreads();

    // ---- GEMM E: (16x256) @ (256x256); 2 col-tiles per wave off one A-frag
    {
        v8f acc0 = {}, acc1 = {};
#pragma unroll
        for (int kc = 0; kc < 8; ++kc) {
            v16bf a  = fragA_lds(sXE + kc * 32, 256, lane);
            v16bf bA = fragB_wt(N1T, 256, wv * 16,        kc * 32, lane);
            v16bf bB = fragB_wt(N1T, 256, (wv + 8) * 16,  kc * 32, lane);
            acc0 = WMMA_BF16(a, bA, acc0);
            acc1 = WMMA_BF16(a, bB, acc1);
        }
        const int n = lane & 15, hi = (lane & 16) ? 8 : 0;
        const int c0 = wv * 16 + n;
        const int c1 = (wv + 8) * 16 + n;
        const float bb0 = b1[c0], bb1 = b1[c1];
#pragma unroll
        for (int r = 0; r < 8; ++r) {
            sH2[(r + hi) * 256 + c0] = f2bf(silu(acc0[r] + bb0));
            sH2[(r + hi) * 256 + c1] = f2bf(silu(acc1[r] + bb1));
        }
    }
    __syncthreads();

    // ---- GEMM F: (16x256) @ (256x2) + bias (tiny -> VALU dots)
    if (tid < 32) {
        const int nd = tid >> 1, cl = tid & 1;
        float d = b2[cl];
        for (int i = 0; i < 256; ++i)
            d += bf2f(sH2[nd * 256 + i]) * bf2f(N2T[cl * 256 + i]);
        sX4[nd * 2 + cl] = d;
    }
    __syncthreads();

    // ---- node_mu = gate*vec_w1 + s_out*(pos - mc[b]);  segment-sum via atomics
    if (tid < 48) {
        const int loc = tid / 3, d = tid % 3;
        const long gn = node0 + loc;
        const int  b  = bidx[gn];
        const float s = sX4[loc * 2 + 0];
        const float g = sX4[loc * 2 + 1];
        const float mcv = pos[gn * 3 + d] - mc[b * 3 + d];
        atomicAdd(&gacc[b * 3 + d], g * sVw1[loc * 3 + d] + s * mcv);
    }
}

// ---------------------------------------------------------------------------
// Final per-graph L2 norm
// ---------------------------------------------------------------------------
__global__ void norm_kernel(const float* __restrict__ gacc, float* __restrict__ out, int B) {
    int b = blockIdx.x * blockDim.x + threadIdx.x;
    if (b < B) {
        float x = gacc[b * 3 + 0], y = gacc[b * 3 + 1], z = gacc[b * 3 + 2];
        out[b] = sqrtf(x * x + y * y + z * z);
    }
}

// ---------------------------------------------------------------------------
extern "C" void kernel_launch(void* const* d_in, const int* in_sizes, int n_in,
                              void* d_out, int out_size, void* d_ws, size_t ws_size,
                              hipStream_t stream)
{
    (void)n_in; (void)ws_size;
    const float* pos    = (const float*)d_in[0];
    const float* mc     = (const float*)d_in[1];
    const float* scal   = (const float*)d_in[2];
    const float* vec    = (const float*)d_in[3];
    const int*   bidx   = (const int*)d_in[4];
    const float* g0w    = (const float*)d_in[6];
    const float* g0m1w  = (const float*)d_in[7];
    const float* g0m1b  = (const float*)d_in[8];
    const float* g0m2w  = (const float*)d_in[9];
    const float* g0m2b  = (const float*)d_in[10];
    const float* g1w    = (const float*)d_in[11];
    const float* g1m1w  = (const float*)d_in[12];
    const float* g1m1b  = (const float*)d_in[13];
    const float* g1m2w  = (const float*)d_in[14];
    const float* g1m2b  = (const float*)d_in[15];

    const long N = in_sizes[0] / 3;     // pos is (N,3)
    const int  B = out_size;            // one norm per graph

    char* ws = (char*)d_ws;
    size_t off = 0;
    auto alloc = [&](size_t bytes) -> void* {
        void* p = ws + off;
        off += (bytes + 255) & ~(size_t)255;
        return p;
    };
    unsigned short* W0T    = (unsigned short*)alloc(384 * 256 * 2);
    unsigned short* M1T0   = (unsigned short*)alloc(512 * 512 * 2);
    unsigned short* M2T0   = (unsigned short*)alloc(256 * 512 * 2);
    unsigned short* W1T    = (unsigned short*)alloc(128 * 128 * 2);
    float*          w1last = (float*)alloc(128 * 4);
    unsigned short* N1T    = (unsigned short*)alloc(256 * 256 * 2);
    unsigned short* N2T    = (unsigned short*)alloc(2 * 256 * 2);
    float*          gacc   = (float*)alloc((size_t)B * 3 * 4);
    unsigned short* qws    = (unsigned short*)alloc((size_t)N * 128 * 2);
    unsigned short* muws   = (unsigned short*)alloc((size_t)N * 3 * 128 * 2);

    conv_wT_kernel<<<(256 * 384 + 255) / 256, 256, 0, stream>>>(g0w, W0T, 256, 384);
    conv_wT_kernel<<<(512 * 512 + 255) / 256, 256, 0, stream>>>(g0m1w, M1T0, 512, 512);
    conv_wT_kernel<<<(512 * 256 + 255) / 256, 256, 0, stream>>>(g0m2w, M2T0, 512, 256);
    conv_g1w_kernel<<<(128 * 129 + 255) / 256, 256, 0, stream>>>(g1w, W1T, w1last);
    conv_wT_kernel<<<(256 * 256 + 255) / 256, 256, 0, stream>>>(g1m1w, N1T, 256, 256);
    conv_wT_kernel<<<(256 * 2 + 255) / 256, 256, 0, stream>>>(g1m2w, N2T, 256, 2);
    zero_kernel<<<(B * 3 + 255) / 256, 256, 0, stream>>>(gacc, B * 3);

    const int nblocks = (int)(N / 16);
    gate0_kernel<<<nblocks, 256, 0, stream>>>(vec, scal, W0T, M1T0, M2T0,
                                              g0m1b, g0m2b, qws, muws);
    gate1_kernel<<<nblocks, 256, 0, stream>>>(muws, qws, W1T, w1last, N1T, g1m1b,
                                              N2T, g1m2b, pos, mc, bidx, gacc);
    norm_kernel<<<(B + 255) / 256, 256, 0, stream>>>(gacc, (float*)d_out, B);
}
